// Grok1MoE_25005299597374
// MI455X (gfx1250) — matrix operand order repaired and checked by reading, not compiled, not run on hardware
//
#include <hip/hip_runtime.h>

// ---- Problem constants (match reference) ----
#define T_TOK 2048
#define HDIM  1024
#define NEXP  8
#define IDIM  2048
#define NSLOT (T_TOK * 2)       // total routed (token, expert) pairs = T*TOP_K
#define SOFTCAP_INV (1.0f / 30.0f)

typedef _Float16 half4  __attribute__((ext_vector_type(4)));
typedef _Float16 half8  __attribute__((ext_vector_type(8)));
typedef _Float16 half16 __attribute__((ext_vector_type(16)));
typedef float    f32x8  __attribute__((ext_vector_type(8)));

// ---------------- WMMA fragment loaders (per CDNA5 ISA layouts) ----------------
// A 16-bit 16x32: lane<16 holds M=lane, K = {0..7, 16..23}; lane>=16: M=lane-16, K={8..15, 24..31}
__device__ __forceinline__ half16 frag_a_lds(const _Float16* row, int lane) {
    const int k0 = (lane & 16) ? 8 : 0;
    union { half16 v; half8 h[2]; } u;
    u.h[0] = *(const half8*)(row + k0);
    u.h[1] = *(const half8*)(row + k0 + 16);
    return u.v;
}
// B 16-bit 32x16 (stored transposed in LDS as Bt[n][k]):
// lane<16 holds N=lane, K={0..7, 16..23}; lane>=16 holds N=lane-16, K={8..15, 24..31}
__device__ __forceinline__ half16 frag_b_lds(const _Float16* row, int lane) {
    const int k0 = (lane & 16) ? 8 : 0;
    union { half16 v; half8 h[2]; } u;
    u.h[0] = *(const half8*)(row + k0);
    u.h[1] = *(const half8*)(row + k0 + 16);
    return u.v;
}

__device__ __forceinline__ float gelu_tanh(float x) {
    const float c0 = 0.7978845608028654f;   // sqrt(2/pi)
    const float c1 = 0.044715f;
    float t = tanhf(c0 * (x + c1 * x * x * x));
    return 0.5f * x * (1.0f + t);
}

// ---------------- Kernel 0: zero metadata ----------------
__global__ void zero_meta_kernel(int* counts, int* cursors) {
    int i = threadIdx.x;
    if (i < NEXP) { counts[i] = 0; cursors[i] = 0; }
}

// ---------------- Kernel 1: router (wave per token) ----------------
__global__ __launch_bounds__(256) void router_kernel(
    const float* __restrict__ x, const float* __restrict__ gate,
    int* __restrict__ tok_e, float* __restrict__ tok_p, int* __restrict__ counts) {
    const int wave = threadIdx.x >> 5;
    const int lane = threadIdx.x & 31;
    const int t = blockIdx.x * 8 + wave;

    float acc[NEXP];
#pragma unroll
    for (int e = 0; e < NEXP; ++e) acc[e] = 0.0f;

    for (int h = lane; h < HDIM; h += 32) {
        float xv = x[(size_t)t * HDIM + h];
        const float4* g4 = (const float4*)(gate + (size_t)h * NEXP);
        float4 a = g4[0], b = g4[1];
        acc[0] += xv * a.x; acc[1] += xv * a.y; acc[2] += xv * a.z; acc[3] += xv * a.w;
        acc[4] += xv * b.x; acc[5] += xv * b.y; acc[6] += xv * b.z; acc[7] += xv * b.w;
    }
#pragma unroll
    for (int off = 16; off > 0; off >>= 1)
#pragma unroll
        for (int e = 0; e < NEXP; ++e) acc[e] += __shfl_xor(acc[e], off, 32);

    if (lane == 0) {
        float v[NEXP];
#pragma unroll
        for (int e = 0; e < NEXP; ++e) v[e] = tanhf(acc[e] * SOFTCAP_INV);
        int e0 = 0;
#pragma unroll
        for (int e = 1; e < NEXP; ++e) if (v[e] > v[e0]) e0 = e;
        int e1 = (e0 == 0) ? 1 : 0;
#pragma unroll
        for (int e = 0; e < NEXP; ++e) if (e != e0 && v[e] > v[e1]) e1 = e;
        float m = fmaxf(v[e0], v[e1]);
        float x0 = expf(v[e0] - m), x1 = expf(v[e1] - m);
        float inv = 1.0f / (x0 + x1);
        tok_e[t * 2 + 0] = e0;  tok_e[t * 2 + 1] = e1;
        tok_p[t * 2 + 0] = x0 * inv;  tok_p[t * 2 + 1] = x1 * inv;
        atomicAdd(&counts[e0], 1);
        atomicAdd(&counts[e1], 1);
    }
}

// ---------------- Kernel 2: scan (8 experts) ----------------
__global__ void scan_kernel(const int* __restrict__ counts, int* __restrict__ offsets) {
    int s = 0;
    offsets[0] = 0;
    for (int e = 0; e < NEXP; ++e) { s += counts[e]; offsets[e + 1] = s; }
}

// ---------------- Kernel 3: assign compacted slots ----------------
__global__ void assign_kernel(const int* __restrict__ tok_e, const float* __restrict__ tok_p,
                              const int* __restrict__ offsets, int* __restrict__ cursors,
                              int* __restrict__ slot_token, float* __restrict__ slot_prob,
                              int* __restrict__ token_slots) {
    int t = blockIdx.x * blockDim.x + threadIdx.x;
    if (t >= T_TOK) return;
#pragma unroll
    for (int k = 0; k < 2; ++k) {
        int e = tok_e[t * 2 + k];
        int s = offsets[e] + atomicAdd(&cursors[e], 1);
        slot_token[s] = t;
        slot_prob[s] = tok_p[t * 2 + k];
        token_slots[t * 2 + k] = s;
    }
}

// ---------------- Kernel 4: gather routed rows -> f16 (vectorized) ----------------
__global__ __launch_bounds__(256) void gather_kernel(
    const float* __restrict__ x, const int* __restrict__ slot_token, _Float16* __restrict__ Xg) {
    int s = blockIdx.x;
    int t = slot_token[s];
    int c4 = threadIdx.x * 4;                       // 256 thr * 4 = 1024 = HDIM
    float4 w = *(const float4*)(x + (size_t)t * HDIM + c4);
    half4 h;
    h[0] = (_Float16)w.x; h[1] = (_Float16)w.y; h[2] = (_Float16)w.z; h[3] = (_Float16)w.w;
    *(half4*)(Xg + (size_t)s * HDIM + c4) = h;
}

// ---------------- Kernel 5: fused GEMM1 (G = X*wg, U = X*wu, act = gelu(G)*U) ----------------
// Block 256 thr (8 waves). Tile M=128, N=64. Wave grid 4(M)x2(N); wave tile 32x32 (2x2 WMMA).
// Double-buffered LDS + register-staged global loads to overlap HBM with WMMA.
__global__ __launch_bounds__(256) void gemm1_kernel(
    const _Float16* __restrict__ Xg, const float* __restrict__ wg, const float* __restrict__ wu,
    const int* __restrict__ offsets, _Float16* __restrict__ Act) {
    const int e = blockIdx.z;
    const int off = offsets[e];
    const int cnt = offsets[e + 1] - off;
    const int m0 = blockIdx.y * 128;
    if (m0 >= cnt) return;
    const int n0 = blockIdx.x * 64;
    const float* wgE = wg + (size_t)e * HDIM * IDIM;
    const float* wuE = wu + (size_t)e * HDIM * IDIM;

    __shared__ _Float16 As[2][128][40];   // M x K (stride 40 keeps 16B alignment)
    __shared__ _Float16 Bg[2][64][40];    // N x K (transposed)
    __shared__ _Float16 Bu[2][64][40];

    const int tid  = threadIdx.x;
    const int lane = tid & 31;
    const int wave = tid >> 5;
    const int wm = wave >> 1;     // 0..3 : 32-row strip
    const int wn = wave & 1;      // 0..1 : 32-col strip
    const int lr = lane & 15;

    const f32x8 zf = {0.f, 0.f, 0.f, 0.f, 0.f, 0.f, 0.f, 0.f};
    f32x8 accg[2][2], accu[2][2];
#pragma unroll
    for (int i = 0; i < 2; ++i)
#pragma unroll
        for (int j = 0; j < 2; ++j) { accg[i][j] = zf; accu[i][j] = zf; }

    // Stage mappings
    const int ar = tid >> 1;            // 0..127 (A row)
    const int ac = (tid & 1) * 16;      // 0 or 16 (A col base, 2x half8)
    const int bn4 = (tid & 15) * 4;     // 0..60  (B col base, float4)
    const int bk0 = tid >> 4;           // 0..15  (B k row; +16 for second)

    const half8 zh = {(_Float16)0, (_Float16)0, (_Float16)0, (_Float16)0,
                      (_Float16)0, (_Float16)0, (_Float16)0, (_Float16)0};

    half8 aS0, aS1;
    float4 bgS0, bgS1, buS0, buS1;

    auto loadStage = [&](int kb) {
        aS0 = zh; aS1 = zh;
        if (m0 + ar < cnt) {
            const _Float16* ap = Xg + (size_t)(off + m0 + ar) * HDIM + kb + ac;
            aS0 = *(const half8*)(ap);
            aS1 = *(const half8*)(ap + 8);
        }
        const float* gp = wgE + (size_t)(kb + bk0) * IDIM + n0 + bn4;
        bgS0 = *(const float4*)(gp);
        bgS1 = *(const float4*)(gp + (size_t)16 * IDIM);
        const float* up = wuE + (size_t)(kb + bk0) * IDIM + n0 + bn4;
        buS0 = *(const float4*)(up);
        buS1 = *(const float4*)(up + (size_t)16 * IDIM);
    };
    auto storeStage = [&](int buf) {
        *(half8*)&As[buf][ar][ac]     = aS0;
        *(half8*)&As[buf][ar][ac + 8] = aS1;
        Bg[buf][bn4 + 0][bk0] = (_Float16)bgS0.x;
        Bg[buf][bn4 + 1][bk0] = (_Float16)bgS0.y;
        Bg[buf][bn4 + 2][bk0] = (_Float16)bgS0.z;
        Bg[buf][bn4 + 3][bk0] = (_Float16)bgS0.w;
        Bg[buf][bn4 + 0][bk0 + 16] = (_Float16)bgS1.x;
        Bg[buf][bn4 + 1][bk0 + 16] = (_Float16)bgS1.y;
        Bg[buf][bn4 + 2][bk0 + 16] = (_Float16)bgS1.z;
        Bg[buf][bn4 + 3][bk0 + 16] = (_Float16)bgS1.w;
        Bu[buf][bn4 + 0][bk0] = (_Float16)buS0.x;
        Bu[buf][bn4 + 1][bk0] = (_Float16)buS0.y;
        Bu[buf][bn4 + 2][bk0] = (_Float16)buS0.z;
        Bu[buf][bn4 + 3][bk0] = (_Float16)buS0.w;
        Bu[buf][bn4 + 0][bk0 + 16] = (_Float16)buS1.x;
        Bu[buf][bn4 + 1][bk0 + 16] = (_Float16)buS1.y;
        Bu[buf][bn4 + 2][bk0 + 16] = (_Float16)buS1.z;
        Bu[buf][bn4 + 3][bk0 + 16] = (_Float16)buS1.w;
    };

    loadStage(0);
    storeStage(0);
    __syncthreads();

    int buf = 0;
    for (int kb = 0; kb < HDIM; kb += 32) {
        const bool nxt = (kb + 32) < HDIM;
        if (nxt) {
            loadStage(kb + 32);
            if (kb + 64 < HDIM) {
                __builtin_prefetch(wgE + (size_t)(kb + 64 + bk0) * IDIM + n0 + bn4, 0, 1);
                __builtin_prefetch(wuE + (size_t)(kb + 64 + bk0) * IDIM + n0 + bn4, 0, 1);
            }
        }
        // Compute on current buffer while staged global loads are in flight
        half16 a0  = frag_a_lds(&As[buf][wm * 32 + lr][0], lane);
        half16 a1  = frag_a_lds(&As[buf][wm * 32 + 16 + lr][0], lane);
        half16 bg0 = frag_b_lds(&Bg[buf][wn * 32 + lr][0], lane);
        half16 bg1 = frag_b_lds(&Bg[buf][wn * 32 + 16 + lr][0], lane);
        half16 bu0 = frag_b_lds(&Bu[buf][wn * 32 + lr][0], lane);
        half16 bu1 = frag_b_lds(&Bu[buf][wn * 32 + 16 + lr][0], lane);

        accg[0][0] = __builtin_amdgcn_wmma_f32_16x16x32_f16(false, a0, false, bg0, (short)0, accg[0][0], false, false);
        accg[0][1] = __builtin_amdgcn_wmma_f32_16x16x32_f16(false, a0, false, bg1, (short)0, accg[0][1], false, false);
        accg[1][0] = __builtin_amdgcn_wmma_f32_16x16x32_f16(false, a1, false, bg0, (short)0, accg[1][0], false, false);
        accg[1][1] = __builtin_amdgcn_wmma_f32_16x16x32_f16(false, a1, false, bg1, (short)0, accg[1][1], false, false);
        accu[0][0] = __builtin_amdgcn_wmma_f32_16x16x32_f16(false, a0, false, bu0, (short)0, accu[0][0], false, false);
        accu[0][1] = __builtin_amdgcn_wmma_f32_16x16x32_f16(false, a0, false, bu1, (short)0, accu[0][1], false, false);
        accu[1][0] = __builtin_amdgcn_wmma_f32_16x16x32_f16(false, a1, false, bu0, (short)0, accu[1][0], false, false);
        accu[1][1] = __builtin_amdgcn_wmma_f32_16x16x32_f16(false, a1, false, bu1, (short)0, accu[1][1], false, false);

        if (nxt) storeStage(buf ^ 1);
        __syncthreads();
        buf ^= 1;
    }

    // Epilogue: act = gelu(g) * u -> f16
    const int hi = (lane & 16) ? 8 : 0;
#pragma unroll
    for (int fm = 0; fm < 2; ++fm)
#pragma unroll
        for (int fn = 0; fn < 2; ++fn)
#pragma unroll
            for (int r = 0; r < 8; ++r) {
                int row = m0 + wm * 32 + fm * 16 + r + hi;
                if (row < cnt) {
                    int col = n0 + wn * 32 + fn * 16 + lr;
                    float g = accg[fm][fn][r];
                    float u = accu[fm][fn][r];
                    Act[(size_t)(off + row) * IDIM + col] = (_Float16)(gelu_tanh(g) * u);
                }
            }
}

// ---------------- Kernel 6: GEMM2 (Y = act * wd) ----------------
// Block 256 thr. Tile M=128, N=128. Wave grid 2(M)x4(N); wave tile 64x32 (4x2 WMMA).
__global__ __launch_bounds__(256) void gemm2_kernel(
    const _Float16* __restrict__ Act, const float* __restrict__ wd,
    const int* __restrict__ offsets, float* __restrict__ Y) {
    const int e = blockIdx.z;
    const int off = offsets[e];
    const int cnt = offsets[e + 1] - off;
    const int m0 = blockIdx.y * 128;
    if (m0 >= cnt) return;
    const int n0 = blockIdx.x * 128;
    const float* wdE = wd + (size_t)e * IDIM * HDIM;

    __shared__ _Float16 As[2][128][40];
    __shared__ _Float16 Bt[2][128][40];

    const int tid  = threadIdx.x;
    const int lane = tid & 31;
    const int wave = tid >> 5;
    const int wm = wave >> 2;     // 0..1 : 64-row strip
    const int wn = wave & 3;      // 0..3 : 32-col strip
    const int lr = lane & 15;

    const f32x8 zf = {0.f, 0.f, 0.f, 0.f, 0.f, 0.f, 0.f, 0.f};
    f32x8 acc[4][2];
#pragma unroll
    for (int i = 0; i < 4; ++i)
#pragma unroll
        for (int j = 0; j < 2; ++j) acc[i][j] = zf;

    const int ar = tid >> 1;
    const int ac = (tid & 1) * 16;
    const int bn4 = (tid & 31) * 4;     // 0..124
    const int bk0 = tid >> 5;           // 0..7 ; rows bk0, +8, +16, +24

    const half8 zh = {(_Float16)0, (_Float16)0, (_Float16)0, (_Float16)0,
                      (_Float16)0, (_Float16)0, (_Float16)0, (_Float16)0};

    half8 aS0, aS1;
    float4 bS0, bS1, bS2, bS3;

    auto loadStage = [&](int kb) {
        aS0 = zh; aS1 = zh;
        if (m0 + ar < cnt) {
            const _Float16* ap = Act + (size_t)(off + m0 + ar) * IDIM + kb + ac;
            aS0 = *(const half8*)(ap);
            aS1 = *(const half8*)(ap + 8);
        }
        const float* bp = wdE + (size_t)(kb + bk0) * HDIM + n0 + bn4;
        bS0 = *(const float4*)(bp);
        bS1 = *(const float4*)(bp + (size_t)8 * HDIM);
        bS2 = *(const float4*)(bp + (size_t)16 * HDIM);
        bS3 = *(const float4*)(bp + (size_t)24 * HDIM);
    };
    auto storeStage = [&](int buf) {
        *(half8*)&As[buf][ar][ac]     = aS0;
        *(half8*)&As[buf][ar][ac + 8] = aS1;
        Bt[buf][bn4 + 0][bk0] = (_Float16)bS0.x;
        Bt[buf][bn4 + 1][bk0] = (_Float16)bS0.y;
        Bt[buf][bn4 + 2][bk0] = (_Float16)bS0.z;
        Bt[buf][bn4 + 3][bk0] = (_Float16)bS0.w;
        Bt[buf][bn4 + 0][bk0 + 8] = (_Float16)bS1.x;
        Bt[buf][bn4 + 1][bk0 + 8] = (_Float16)bS1.y;
        Bt[buf][bn4 + 2][bk0 + 8] = (_Float16)bS1.z;
        Bt[buf][bn4 + 3][bk0 + 8] = (_Float16)bS1.w;
        Bt[buf][bn4 + 0][bk0 + 16] = (_Float16)bS2.x;
        Bt[buf][bn4 + 1][bk0 + 16] = (_Float16)bS2.y;
        Bt[buf][bn4 + 2][bk0 + 16] = (_Float16)bS2.z;
        Bt[buf][bn4 + 3][bk0 + 16] = (_Float16)bS2.w;
        Bt[buf][bn4 + 0][bk0 + 24] = (_Float16)bS3.x;
        Bt[buf][bn4 + 1][bk0 + 24] = (_Float16)bS3.y;
        Bt[buf][bn4 + 2][bk0 + 24] = (_Float16)bS3.z;
        Bt[buf][bn4 + 3][bk0 + 24] = (_Float16)bS3.w;
    };

    loadStage(0);
    storeStage(0);
    __syncthreads();

    int buf = 0;
    for (int kb = 0; kb < IDIM; kb += 32) {
        const bool nxt = (kb + 32) < IDIM;
        if (nxt) {
            loadStage(kb + 32);
            if (kb + 64 < IDIM)
                __builtin_prefetch(wdE + (size_t)(kb + 64 + bk0) * HDIM + n0 + bn4, 0, 1);
        }

        half16 b0 = frag_b_lds(&Bt[buf][wn * 32 + lr][0], lane);
        half16 b1 = frag_b_lds(&Bt[buf][wn * 32 + 16 + lr][0], lane);
#pragma unroll
        for (int fm = 0; fm < 4; ++fm) {
            half16 a = frag_a_lds(&As[buf][wm * 64 + fm * 16 + lr][0], lane);
            acc[fm][0] = __builtin_amdgcn_wmma_f32_16x16x32_f16(false, a, false, b0, (short)0, acc[fm][0], false, false);
            acc[fm][1] = __builtin_amdgcn_wmma_f32_16x16x32_f16(false, a, false, b1, (short)0, acc[fm][1], false, false);
        }

        if (nxt) storeStage(buf ^ 1);
        __syncthreads();
        buf ^= 1;
    }

    const int hi = (lane & 16) ? 8 : 0;
#pragma unroll
    for (int fm = 0; fm < 4; ++fm)
#pragma unroll
        for (int fn = 0; fn < 2; ++fn)
#pragma unroll
            for (int r = 0; r < 8; ++r) {
                int row = m0 + wm * 64 + fm * 16 + r + hi;
                if (row < cnt) {
                    int col = n0 + wn * 32 + fn * 16 + lr;
                    Y[(size_t)(off + row) * HDIM + col] = acc[fm][fn][r];
                }
            }
}

// ---------------- Kernel 7: weighted combine (deterministic, vectorized) ----------------
__global__ __launch_bounds__(256) void combine_kernel(
    const float* __restrict__ Y, const int* __restrict__ token_slots,
    const float* __restrict__ slot_prob, float* __restrict__ out) {
    int idx = blockIdx.x * blockDim.x + threadIdx.x;   // over T*H/4
    int t  = idx >> 8;                                  // H/4 = 256 vec4 per token
    int h4 = (idx & 255) * 4;
    int s0 = token_slots[t * 2 + 0];
    int s1 = token_slots[t * 2 + 1];
    float p0 = slot_prob[s0], p1 = slot_prob[s1];
    float4 y0 = *(const float4*)(Y + (size_t)s0 * HDIM + h4);
    float4 y1 = *(const float4*)(Y + (size_t)s1 * HDIM + h4);
    float4 o;
    o.x = p0 * y0.x + p1 * y1.x;
    o.y = p0 * y0.y + p1 * y1.y;
    o.z = p0 * y0.z + p1 * y1.z;
    o.w = p0 * y0.w + p1 * y1.w;
    *(float4*)(out + (size_t)t * HDIM + h4) = o;
}

// ---------------- Launch ----------------
extern "C" void kernel_launch(void* const* d_in, const int* in_sizes, int n_in,
                              void* d_out, int out_size, void* d_ws, size_t ws_size,
                              hipStream_t stream) {
    const float* x    = (const float*)d_in[0];   // [T,H]
    const float* gate = (const float*)d_in[1];   // [H,E]
    const float* wg   = (const float*)d_in[2];   // [E,H,I]
    const float* wu   = (const float*)d_in[3];   // [E,H,I]
    const float* wd   = (const float*)d_in[4];   // [E,I,H]
    float* out = (float*)d_out;                  // [T,H]

    char* p = (char*)d_ws;
    auto alloc = [&](size_t bytes) -> char* {
        char* r = p;
        p += (bytes + 255) & ~(size_t)255;
        return r;
    };
    int*   counts      = (int*)alloc(NEXP * sizeof(int));
    int*   cursors     = (int*)alloc(NEXP * sizeof(int));
    int*   offsets     = (int*)alloc((NEXP + 1) * sizeof(int));
    int*   tok_e       = (int*)alloc((size_t)T_TOK * 2 * sizeof(int));
    float* tok_p       = (float*)alloc((size_t)T_TOK * 2 * sizeof(float));
    int*   slot_token  = (int*)alloc((size_t)NSLOT * sizeof(int));
    float* slot_prob   = (float*)alloc((size_t)NSLOT * sizeof(float));
    int*   token_slots = (int*)alloc((size_t)T_TOK * 2 * sizeof(int));
    _Float16* Xg  = (_Float16*)alloc((size_t)NSLOT * HDIM * sizeof(_Float16));   // 8 MiB
    _Float16* Act = (_Float16*)alloc((size_t)NSLOT * IDIM * sizeof(_Float16));   // 16 MiB
    float*    Y   = (float*)alloc((size_t)NSLOT * HDIM * sizeof(float));         // 16 MiB

    zero_meta_kernel<<<1, 32, 0, stream>>>(counts, cursors);
    router_kernel<<<T_TOK / 8, 256, 0, stream>>>(x, gate, tok_e, tok_p, counts);
    scan_kernel<<<1, 1, 0, stream>>>(counts, offsets);
    assign_kernel<<<T_TOK / 256, 256, 0, stream>>>(tok_e, tok_p, offsets, cursors,
                                                   slot_token, slot_prob, token_slots);
    gather_kernel<<<NSLOT, 256, 0, stream>>>(x, slot_token, Xg);

    dim3 g1(IDIM / 64, T_TOK / 128, NEXP);   // (32, 16, 8); blocks beyond cnt exit early
    gemm1_kernel<<<g1, 256, 0, stream>>>(Xg, wg, wu, offsets, Act);

    dim3 g2(HDIM / 128, T_TOK / 128, NEXP);  // (8, 16, 8)
    gemm2_kernel<<<g2, 256, 0, stream>>>(Act, wd, offsets, Y);

    combine_kernel<<<(T_TOK * HDIM / 4) / 256, 256, 0, stream>>>(Y, token_slots, slot_prob, out);
}
